// CrossAttention_7370163880642
// MI455X (gfx1250) — compile-verified, matrix operand order checked by reading, not run on hardware
//
#include <hip/hip_runtime.h>
#include <hip/hip_bf16.h>
#include <math.h>
#include <stdint.h>

typedef __attribute__((ext_vector_type(16))) _Float16 v16h;
typedef __attribute__((ext_vector_type(8)))  _Float16 v8h;
typedef __attribute__((ext_vector_type(4)))  _Float16 v4h;
typedef __attribute__((ext_vector_type(8)))  float    v8f;
typedef __attribute__((ext_vector_type(4)))  int      v4i;

#ifndef __has_builtin
#define __has_builtin(x) 0
#endif

#if __has_builtin(__builtin_amdgcn_global_load_async_to_lds_b128)
#define USE_ASYNC_LDS 1
#else
#define USE_ASYNC_LDS 0
#endif

#if USE_ASYNC_LDS
// GLOBAL_LOAD_ASYNC_TO_LDS_B128: 16B global -> LDS, tracked by ASYNCcnt.
// Builtin signature (from hipcc diagnostic): (int4 AS1* src, int4 AS3* dst,
// imm offset, imm cpol).
__device__ __forceinline__ void async_b128(const _Float16* g, _Float16* l) {
  typedef v4i __attribute__((address_space(1)))* gp_t;
  typedef v4i __attribute__((address_space(3)))* lp_t;
  __builtin_amdgcn_global_load_async_to_lds_b128(
      (gp_t)(uintptr_t)g, (lp_t)(uintptr_t)l, 0, 0);
}
__device__ __forceinline__ void wait_async0() {
#if __has_builtin(__builtin_amdgcn_s_wait_asynccnt)
  __builtin_amdgcn_s_wait_asynccnt(0);
#else
  asm volatile("s_wait_asynccnt 0x0" ::: "memory");
#endif
}
#endif

// D = A(16x32 f16) * B(32x16 f16) + C(16x16 f32)
__device__ __forceinline__ v8f wmma_f16(v16h a, v16h b, v8f c) {
  return __builtin_amdgcn_wmma_f32_16x16x32_f16(false, a, false, b, (short)0, c,
                                                false, false);
}

// Max-reduce across the 16 lanes of a DPP row (lanes 0-15 / 16-31 independent)
// using a VALU-only butterfly: xor1, xor2 (quad_perm), half-mirror, mirror.
__device__ __forceinline__ float row_max16(float x) {
  x = fmaxf(x, __int_as_float(__builtin_amdgcn_mov_dpp(
                   __float_as_int(x), 0xB1, 0xF, 0xF, true)));  // quad xor1
  x = fmaxf(x, __int_as_float(__builtin_amdgcn_mov_dpp(
                   __float_as_int(x), 0x4E, 0xF, 0xF, true)));  // quad xor2
  x = fmaxf(x, __int_as_float(__builtin_amdgcn_mov_dpp(
                   __float_as_int(x), 0x141, 0xF, 0xF, true))); // half mirror
  x = fmaxf(x, __int_as_float(__builtin_amdgcn_mov_dpp(
                   __float_as_int(x), 0x140, 0xF, 0xF, true))); // row mirror
  return x;
}

// Load one 16x32 f16 operand fragment (A layout; B uses same layout on the
// transposed-storage tile). Per ISA 7.12.2: lane L holds row (L&15), k-chunks
// {0..7,16..23} for L<16 and {8..15,24..31} for L>=16.
// stride must be a multiple of 8 halfs (16B) for aligned b128 accesses.
template <typename T>
__device__ __forceinline__ v16h load_frag(const T* base, int stride) {
  const int lane = threadIdx.x & 31;
  const int row  = lane & 15;
  const int koff = (lane >> 4) << 3;  // 0 or 8
  const T* p = base + row * stride + koff;
  v16h f;
  *(v8h*)&f         = *(const v8h*)(p);        // k = koff .. koff+7
  *(((v8h*)&f) + 1) = *(const v8h*)(p + 16);   // k = koff+16 .. koff+23
  return f;
}

// ---------------------------------------------------------------------------
// Projection GEMM: C[M=8192, N] = A[M,1024] * W[1024,N] + bias
// MODE 0: A=f32 x_q,  N=1024 -> Qh f16 [B,H,NQ,DH]
// MODE 1: A=f32 x_kv, N=2048 -> Kh f16 [B,H,NK,DH], Vt f16 [B,H,DH,NK]
// MODE 2: A=f16 O,    N=1024 -> out f32 = resid + C + bias (nan_to_num)
// Block: 256 thr (8 waves), tile 128x128, each wave 32x64 (2x4 WMMA tiles).
// ---------------------------------------------------------------------------
template <int MODE>
__global__ __launch_bounds__(256) void proj_kernel(
    const float* __restrict__ A32, const _Float16* __restrict__ A16,
    const float* __restrict__ W, const float* __restrict__ bias, int Nw,
    _Float16* __restrict__ Qh, _Float16* __restrict__ Kh,
    _Float16* __restrict__ Vt, float* __restrict__ out,
    const float* __restrict__ resid) {
  constexpr int K = 1024;
  constexpr int LDA = 48;  // 32 + 16 pad, 96B rows (16B multiple)
  constexpr int LDB = 48;
  __shared__ _Float16 As[128 * LDA];
  __shared__ _Float16 Bs[128 * LDB];  // stored transposed: Bs[n][k]

  const int tid  = threadIdx.x;
  const int lane = tid & 31;
  const int wave = tid >> 5;
  const int wm = wave & 3;   // 0..3 -> M offset (x32)
  const int wn = wave >> 2;  // 0..1 -> N offset (x64)
  const int m0 = blockIdx.x * 128;
  const int n0 = blockIdx.y * 128;

  v8f acc[2][4] = {};

  for (int k0 = 0; k0 < K; k0 += 32) {
    __syncthreads();
    // ---- stage A tile 128x32 into LDS (convert f32->f16 if needed) ----
    if constexpr (MODE == 2) {
#pragma unroll
      for (int i = 0; i < 2; ++i) {
        int idx = tid + i * 256;       // v8h chunk 0..511
        int row = idx >> 2;
        int col = (idx & 3) << 3;
#if USE_ASYNC_LDS
        async_b128(&A16[(size_t)(m0 + row) * K + k0 + col], &As[row * LDA + col]);
#else
        *(v8h*)&As[row * LDA + col] =
            *(const v8h*)&A16[(size_t)(m0 + row) * K + k0 + col];
#endif
      }
    } else {
#pragma unroll
      for (int i = 0; i < 4; ++i) {
        int idx = (tid + i * 256) << 2;  // element 0..4095
        int row = idx >> 5;
        int col = idx & 31;
        const float4 v = *(const float4*)&A32[(size_t)(m0 + row) * K + k0 + col];
        v4h hv = {(_Float16)v.x, (_Float16)v.y, (_Float16)v.z, (_Float16)v.w};
        *(v4h*)&As[row * LDA + col] = hv;
      }
    }
    // ---- stage B tile transposed: Bs[n][k] = W[k0+k][n0+n], 32x128 ----
#pragma unroll
    for (int i = 0; i < 4; ++i) {
      int idx = (tid + i * 256) << 2;  // element 0..4095
      int k = idx >> 7;
      int n = idx & 127;
      const float4 v = *(const float4*)&W[(size_t)(k0 + k) * Nw + n0 + n];
      Bs[(n + 0) * LDB + k] = (_Float16)v.x;
      Bs[(n + 1) * LDB + k] = (_Float16)v.y;
      Bs[(n + 2) * LDB + k] = (_Float16)v.z;
      Bs[(n + 3) * LDB + k] = (_Float16)v.w;
    }
    // ---- L2 prefetch hints for next k-step (global_prefetch_b8) ----
    if (k0 + 32 < K) {
      int idx = tid << 2;
      if constexpr (MODE == 2) {
        __builtin_prefetch(
            &A16[(size_t)(m0 + (tid >> 2)) * K + (k0 + 32) + ((tid & 3) << 3)],
            0, 0);
      } else {
        __builtin_prefetch(
            &A32[(size_t)(m0 + (idx >> 5)) * K + (k0 + 32) + (idx & 31)], 0, 0);
      }
      __builtin_prefetch(&W[(size_t)(k0 + 32 + (idx >> 7)) * Nw + n0 + (idx & 127)],
                         0, 0);
    }
#if USE_ASYNC_LDS
    if constexpr (MODE == 2) wait_async0();
#endif
    __syncthreads();

    v16h a0 = load_frag(&As[(wm * 32 + 0) * LDA], LDA);
    v16h a1 = load_frag(&As[(wm * 32 + 16) * LDA], LDA);
#pragma unroll
    for (int nt = 0; nt < 4; ++nt) {
      v16h b = load_frag(&Bs[(wn * 64 + nt * 16) * LDB], LDB);
      acc[0][nt] = wmma_f16(a0, b, acc[0][nt]);
      acc[1][nt] = wmma_f16(a1, b, acc[1][nt]);
    }
  }

  // ---- store: C layout lane L, vgpr r -> (row r+8*(L>>4), col L&15) ----
#pragma unroll
  for (int mt = 0; mt < 2; ++mt)
#pragma unroll
    for (int nt = 0; nt < 4; ++nt)
#pragma unroll
      for (int r = 0; r < 8; ++r) {
        int mg = m0 + wm * 32 + mt * 16 + r + ((lane >> 4) << 3);
        int ng = n0 + wn * 64 + nt * 16 + (lane & 15);
        float c = acc[mt][nt][r] + bias[ng];
        if constexpr (MODE == 0) {
          int b = mg >> 11, q = mg & 2047;
          int h = ng >> 6, d = ng & 63;
          Qh[(((size_t)b * 16 + h) * 2048 + q) * 64 + d] = (_Float16)c;
        } else if constexpr (MODE == 1) {
          int b = mg >> 11, t = mg & 2047;
          if (ng < 1024) {
            int h = ng >> 6, d = ng & 63;
            Kh[(((size_t)b * 16 + h) * 2048 + t) * 64 + d] = (_Float16)c;
          } else {
            int n2 = ng - 1024;
            int h = n2 >> 6, d = n2 & 63;
            Vt[(((size_t)b * 16 + h) * 64 + d) * 2048 + t] = (_Float16)c;
          }
        } else {
          float v = resid[(size_t)mg * 1024 + ng] + c;
          v = (v != v) ? 0.0f
                       : fminf(fmaxf(v, -3.4028234664e38f), 3.4028234664e38f);
          out[(size_t)mg * 1024 + ng] = v;
        }
      }
}

// ---------------------------------------------------------------------------
// Fused flash attention: per (b,h), 8 waves x 16 query rows = 128 q / block.
// Streams 64-key K/V tiles through LDS (async-to-LDS when available);
// online softmax. Row max -> DPP16 butterfly (VALU only). Row sums -> extra
// WMMA against an all-ones B fragment: the denominator rides in a 5th
// accumulator, replicated per lane, rescaled by the same corr as O.
// ---------------------------------------------------------------------------
__global__ __launch_bounds__(256) void attn_kernel(
    const _Float16* __restrict__ Qh, const _Float16* __restrict__ Kh,
    const _Float16* __restrict__ Vt, _Float16* __restrict__ Og) {
  constexpr int LDK = 80;  // 64 + 16 pad (160B rows, 16B multiple)
  constexpr int LDV = 80;
  constexpr int LDP = 80;
  __shared__ _Float16 Ks[64 * LDK];       // 64 keys x DH=64
  __shared__ _Float16 Vs[64 * LDV];       // Vt: DH=64 x 64 keys
  __shared__ _Float16 Ps[8 * 16 * LDP];   // per-wave P staging (16 x 64)

  const int tid = threadIdx.x;
  const int lane = tid & 31;
  const int wave = tid >> 5;
  const int bh = blockIdx.x;                  // b*16 + h
  const int q0 = blockIdx.y * 128 + wave * 16;

  // Q fragments for this wave's 16 rows (DH=64 -> two K=32 chunks), from HBM.
  const _Float16* Qbase = Qh + ((size_t)bh * 2048 + q0) * 64;
  const v16h aq0 = load_frag(Qbase + 0, 64);
  const v16h aq1 = load_frag(Qbase + 32, 64);

  v16h ones;
#pragma unroll
  for (int i = 0; i < 16; ++i) ones[i] = (_Float16)1.0f;

  v8f o0 = {}, o1 = {}, o2 = {}, o3 = {}, osum = {};
  float mrow[8];
#pragma unroll
  for (int r = 0; r < 8; ++r) mrow[r] = -1e30f;

  _Float16* Pw = &Ps[wave * 16 * LDP];
  const _Float16* Kg = Kh + (size_t)bh * 2048 * 64;
  const _Float16* Vg = Vt + (size_t)bh * 64 * 2048;

  for (int t0 = 0; t0 < 2048; t0 += 64) {
    __syncthreads();
    // K tile 64x64 and Vt tile 64x64: two 16B chunks per thread each.
#pragma unroll
    for (int i = 0; i < 2; ++i) {
      int idx = tid + i * 256;         // 0..511
      int row = idx >> 3;
      int col = (idx & 7) << 3;
#if USE_ASYNC_LDS
      async_b128(&Kg[(size_t)(t0 + row) * 64 + col], &Ks[row * LDK + col]);
      async_b128(&Vg[(size_t)row * 2048 + t0 + col], &Vs[row * LDV + col]);
#else
      *(v8h*)&Ks[row * LDK + col] =
          *(const v8h*)&Kg[(size_t)(t0 + row) * 64 + col];
      *(v8h*)&Vs[row * LDV + col] =
          *(const v8h*)&Vg[(size_t)row * 2048 + t0 + col];
#endif
    }
#if USE_ASYNC_LDS
    wait_async0();
#endif
    __syncthreads();

    // S = Q * K^T : four 16x16 key tiles, K-accumulate over DH (2 x 32)
    v8f s[4] = {};
#pragma unroll
    for (int nt = 0; nt < 4; ++nt) {
      v16h bk0 = load_frag(&Ks[(nt * 16) * LDK + 0], LDK);
      v16h bk1 = load_frag(&Ks[(nt * 16) * LDK + 32], LDK);
      s[nt] = wmma_f16(aq0, bk0, s[nt]);
      s[nt] = wmma_f16(aq1, bk1, s[nt]);
    }

    // Online softmax per row (rows live in vgpr r across 16 lanes of a half)
#pragma unroll
    for (int r = 0; r < 8; ++r) {
      float e0 = s[0][r] * 0.125f;  // DH^-0.5
      float e1 = s[1][r] * 0.125f;
      float e2 = s[2][r] * 0.125f;
      float e3 = s[3][r] * 0.125f;
      float rmax = row_max16(fmaxf(fmaxf(e0, e1), fmaxf(e2, e3)));
      float mnew = fmaxf(mrow[r], rmax);
      float corr = __expf(mrow[r] - mnew);
      mrow[r] = mnew;
      float p0 = __expf(e0 - mnew);
      float p1 = __expf(e1 - mnew);
      float p2 = __expf(e2 - mnew);
      float p3 = __expf(e3 - mnew);
      o0[r] = o0[r] * corr;
      o1[r] = o1[r] * corr;
      o2[r] = o2[r] * corr;
      o3[r] = o3[r] * corr;
      osum[r] = osum[r] * corr;
      int prow = r + ((lane >> 4) << 3);
      int c = lane & 15;
      Pw[prow * LDP + c]      = (_Float16)p0;
      Pw[prow * LDP + 16 + c] = (_Float16)p1;
      Pw[prow * LDP + 32 + c] = (_Float16)p2;
      Pw[prow * LDP + 48 + c] = (_Float16)p3;
    }

    // O += P(16x64) * V(64x64); denominator += P * ones.
    v16h ap0 = load_frag(Pw + 0, LDP);   // keys 0..31
    v16h ap1 = load_frag(Pw + 32, LDP);  // keys 32..63
    osum = wmma_f16(ap0, ones, osum);
    osum = wmma_f16(ap1, ones, osum);
    {
      v16h bv0 = load_frag(&Vs[0 * LDV + 0], LDV);
      v16h bv1 = load_frag(&Vs[0 * LDV + 32], LDV);
      o0 = wmma_f16(ap0, bv0, o0);
      o0 = wmma_f16(ap1, bv1, o0);
    }
    {
      v16h bv0 = load_frag(&Vs[16 * LDV + 0], LDV);
      v16h bv1 = load_frag(&Vs[16 * LDV + 32], LDV);
      o1 = wmma_f16(ap0, bv0, o1);
      o1 = wmma_f16(ap1, bv1, o1);
    }
    {
      v16h bv0 = load_frag(&Vs[32 * LDV + 0], LDV);
      v16h bv1 = load_frag(&Vs[32 * LDV + 32], LDV);
      o2 = wmma_f16(ap0, bv0, o2);
      o2 = wmma_f16(ap1, bv1, o2);
    }
    {
      v16h bv0 = load_frag(&Vs[48 * LDV + 0], LDV);
      v16h bv1 = load_frag(&Vs[48 * LDV + 32], LDV);
      o3 = wmma_f16(ap0, bv0, o3);
      o3 = wmma_f16(ap1, bv1, o3);
    }
  }

  // Normalize (every lane holds its row's denominator in osum) and store
  // token-major f16 [B, NQ, D] for the output GEMM.
  const int b = bh >> 4, h = bh & 15;
#pragma unroll
  for (int r = 0; r < 8; ++r) {
    float inv = 1.0f / osum[r];
    int qg = q0 + r + ((lane >> 4) << 3);
    size_t base = ((size_t)(b * 2048 + qg)) * 1024 + h * 64 + (lane & 15);
    Og[base + 0]  = (_Float16)(o0[r] * inv);
    Og[base + 16] = (_Float16)(o1[r] * inv);
    Og[base + 32] = (_Float16)(o2[r] * inv);
    Og[base + 48] = (_Float16)(o3[r] * inv);
  }
}

// ---------------------------------------------------------------------------
extern "C" void kernel_launch(void* const* d_in, const int* in_sizes, int n_in,
                              void* d_out, int out_size, void* d_ws,
                              size_t ws_size, hipStream_t stream) {
  const float* x_q  = (const float*)d_in[0];
  const float* x_kv = (const float*)d_in[1];
  const float* Wq   = (const float*)d_in[2];
  const float* bq   = (const float*)d_in[3];
  const float* Wkv  = (const float*)d_in[4];
  const float* bkv  = (const float*)d_in[5];
  const float* Wp   = (const float*)d_in[6];
  const float* bp   = (const float*)d_in[7];
  float* out = (float*)d_out;

  const size_t NELT = (size_t)4 * 16 * 2048 * 64;  // 8,388,608 per buffer
  _Float16* Qh = (_Float16*)d_ws;
  _Float16* Kh = Qh + NELT;
  _Float16* Vt = Kh + NELT;
  _Float16* Og = Vt + NELT;

  // Q projection: M=8192, N=1024
  proj_kernel<0><<<dim3(64, 8), 256, 0, stream>>>(
      x_q, nullptr, Wq, bq, 1024, Qh, Kh, Vt, nullptr, nullptr);
  // KV projection: M=8192, N=2048 (K head-major, V transposed)
  proj_kernel<1><<<dim3(64, 16), 256, 0, stream>>>(
      x_kv, nullptr, Wkv, bkv, 2048, Qh, Kh, Vt, nullptr, nullptr);
  // Flash attention: 64 (b,h) pairs x 16 q-blocks of 128
  attn_kernel<<<dim3(64, 16), 256, 0, stream>>>(Qh, Kh, Vt, Og);
  // Output projection + residual: M=8192, N=1024
  proj_kernel<2><<<dim3(64, 8), 256, 0, stream>>>(
      nullptr, Og, Wp, bp, 1024, Qh, Kh, Vt, out, x_q);
}
